// PsiQRHTransformerBlock_79207786872876
// MI455X (gfx1250) — matrix operand order repaired and checked by reading, not compiled, not run on hardware
//
#include <hip/hip_runtime.h>
#include <hip/hip_bf16.h>
#include <cstdint>

// ---------------------------------------------------------------------------
// CDNA5 (gfx1250) WMMA bf16 pipeline for the PsiQRH transformer block.
// wave32; V_WMMA_F32_16X16X32_BF16; async global->LDS staging (ASYNCcnt)
// with double-buffered tiles when the toolchain exposes the builtins.
// ---------------------------------------------------------------------------

typedef __bf16 bf16_t;
typedef __attribute__((ext_vector_type(16))) __bf16 v16bf;
typedef __attribute__((ext_vector_type(8)))  __bf16 v8bf;
typedef __attribute__((ext_vector_type(8)))  float  v8f;

#ifndef __has_builtin
#define __has_builtin(x) 0
#endif
#if __has_builtin(__builtin_amdgcn_global_load_async_to_lds_b128) && \
    __has_builtin(__builtin_amdgcn_s_wait_asynccnt)
#define USE_ASYNC_LDS 1
// Builtin signature (from hipcc diagnostic): pointers to int vector_size(16),
// global (AS1) source and LDS (AS3) destination, plus imm offset / cpol.
typedef int v4i_vs __attribute__((vector_size(16)));
typedef __attribute__((address_space(1))) v4i_vs* gv4i_p;
typedef __attribute__((address_space(3))) v4i_vs* lv4i_p;
#else
#define USE_ASYNC_LDS 0
#endif

// 16B global -> LDS copy. Async (ASYNCcnt-tracked DMA) when available.
static __device__ __forceinline__ void cp_b128(void* lds_dst, const void* gsrc) {
#if USE_ASYNC_LDS
    __builtin_amdgcn_global_load_async_to_lds_b128((gv4i_p)gsrc, (lv4i_p)lds_dst,
                                                   0, 0);
#else
    *(uint4*)lds_dst = *(const uint4*)gsrc;
#endif
}

static __device__ __forceinline__ void wait_async() {
#if USE_ASYNC_LDS
    __builtin_amdgcn_s_wait_asynccnt(0);
#endif
}

static __device__ inline v16bf make_frag(v8bf lo, v8bf hi) {
    v16bf a;
#pragma unroll
    for (int i = 0; i < 8; ++i) { a[i] = lo[i]; a[i + 8] = hi[i]; }
    return a;
}

static __device__ inline v8f wmma_bf16(v16bf a, v16bf b, v8f c) {
    // (neg_a, A, neg_b, B, c_mod, C, reuse_a, reuse_b)
    return __builtin_amdgcn_wmma_f32_16x16x32_bf16(false, a, false, b,
                                                   (short)0, c, false, false);
}

// ---------------------------------------------------------------------------
// fp32 -> bf16 convert (weights)
// ---------------------------------------------------------------------------
__global__ __launch_bounds__(256)
void cvt_f32_bf16(const float* __restrict__ in, bf16_t* __restrict__ out, int n) {
    int i = blockIdx.x * 256 + threadIdx.x;
    if (i < n) out[i] = (bf16_t)in[i];
}

// ---------------------------------------------------------------------------
// LayerNorm over D=1024, fp32 in -> bf16 out. 128 threads/row.
// ---------------------------------------------------------------------------
__global__ __launch_bounds__(128)
void layernorm_f32_bf16(const float* __restrict__ X, const float* __restrict__ g,
                        const float* __restrict__ b, bf16_t* __restrict__ Y, int D) {
    const int row = blockIdx.x;
    const float* xr = X + (size_t)row * D;
    const int tid = threadIdx.x;
    float vals[8];
    float s = 0.f, s2 = 0.f;
#pragma unroll
    for (int j = 0; j < 8; ++j) {
        float v = xr[tid + j * 128];
        vals[j] = v; s += v; s2 += v * v;
    }
#pragma unroll
    for (int m = 1; m < 32; m <<= 1) {
        s  += __shfl_xor(s,  m, 32);
        s2 += __shfl_xor(s2, m, 32);
    }
    __shared__ float red[8];
    const int wave = tid >> 5, lane = tid & 31;
    if (lane == 0) { red[wave] = s; red[4 + wave] = s2; }
    __syncthreads();
    if (tid == 0) {
        float ts = red[0] + red[1] + red[2] + red[3];
        float t2 = red[4] + red[5] + red[6] + red[7];
        float mean = ts / (float)D;
        float var  = t2 / (float)D - mean * mean;
        red[0] = mean;
        red[1] = rsqrtf(var + 1e-5f);
    }
    __syncthreads();
    const float mean = red[0], rstd = red[1];
    bf16_t* yr = Y + (size_t)row * D;
#pragma unroll
    for (int j = 0; j < 8; ++j) {
        int i = tid + j * 128;
        yr[i] = (bf16_t)((vals[j] - mean) * rstd * g[i] + b[i]);
    }
}

// ---------------------------------------------------------------------------
// WMMA GEMM: C[M,N] = A[M,K] @ W[N,K]^T (+bias). bf16 A/W, fp32 accumulate.
// 128 threads (4 waves), 64x64 tile, K-step 32, double-buffered async LDS.
// MODE 0: fp32 row-major out.
// MODE 1: bf16 d-major head-blocked [B,H,HD,S] out (pre-transposed for attn).
// MODE 2: out = xres + ls * (C + bias)  (fp32, final residual).
// ---------------------------------------------------------------------------
template <int MODE>
__global__ __launch_bounds__(128)
void gemm_wmma_bf16(const bf16_t* __restrict__ A, const bf16_t* __restrict__ W,
                    const float* __restrict__ bias, void* __restrict__ Cout,
                    const float* __restrict__ xres, const float* __restrict__ ls,
                    int M, int N, int K, int S, int H, int HD) {
    __shared__ __align__(16) bf16_t As[2][64][32];
    __shared__ __align__(16) bf16_t Ws[2][64][32];

    const int tid  = threadIdx.x;
    const int wave = tid >> 5;
    const int lane = tid & 31;
    const int half = lane >> 4;     // 0: lanes 0-15, 1: lanes 16-31
    const int ln   = lane & 15;
    const int n0 = blockIdx.x * 64;
    const int m0 = blockIdx.y * 64;
    const int mlocal = wave * 16 + ln;

    auto stageAW = [&](int bb, int k0) {
#pragma unroll
        for (int i = 0; i < 2; ++i) {
            int c = tid + i * 128;
            int r = c >> 2, cc = c & 3;
            cp_b128(&As[bb][r][cc * 8], A + (size_t)(m0 + r) * K + k0 + cc * 8);
            cp_b128(&Ws[bb][r][cc * 8], W + (size_t)(n0 + r) * K + k0 + cc * 8);
        }
    };

    v8f acc[4] = {};

    stageAW(0, 0);
    wait_async();
    __syncthreads();

    int cur = 0;
    for (int k0 = 0; k0 < K; k0 += 32) {
        if (k0 + 32 < K) stageAW(cur ^ 1, k0 + 32);  // overlap next slab

        // A fragment (16x32, M=lane%16, per-lane K split per CDNA5 layout)
        const int b0 = half ? 8 : 0;
        v8bf alo = *(const v8bf*)&As[cur][mlocal][b0];
        v8bf ahi = *(const v8bf*)&As[cur][mlocal][b0 + 16];
        v16bf af = make_frag(alo, ahi);

#pragma unroll
        for (int t = 0; t < 4; ++t) {
            const int nl  = t * 16 + ln;
            const int kk0 = half ? 8 : 0;
            v8bf blo = *(const v8bf*)&Ws[cur][nl][kk0];
            v8bf bhi = *(const v8bf*)&Ws[cur][nl][kk0 + 16];
            acc[t] = wmma_bf16(af, make_frag(blo, bhi), acc[t]);
        }

        wait_async();
        __syncthreads();
        cur ^= 1;
    }

    // Epilogue: D layout -> lanes 0-15 hold rows r, lanes 16-31 rows 8+r.
#pragma unroll
    for (int t = 0; t < 4; ++t) {
        const int n = n0 + t * 16 + ln;
        const float bv = bias ? bias[n] : 0.f;
#pragma unroll
        for (int r = 0; r < 8; ++r) {
            const int m = m0 + wave * 16 + (half ? 8 + r : r);
            const float v = acc[t][r] + bv;
            if (MODE == 0) {
                ((float*)Cout)[(size_t)m * N + n] = v;
            } else if (MODE == 1) {
                // d-major: Hm[b][h][d][s]
                const int bi = m / S, ss = m - bi * S;
                const int hh = n / HD, dd = n - hh * HD;
                ((bf16_t*)Cout)[(((size_t)bi * H + hh) * HD + dd) * S + ss] = (bf16_t)v;
            } else {
                const size_t idx = (size_t)m * N + n;
                ((float*)Cout)[idx] = xres[idx] + ls[n] * v;
            }
        }
    }
}

// ---------------------------------------------------------------------------
// Phasor: phi = Q_head @ phi_w^T + phi_b ; emit bf16 [Q*cos(phi) || Q*sin(phi)]
// into per-head rows of length 2*HD=128. One block of 64 threads per (b,s,h).
// ---------------------------------------------------------------------------
__global__ __launch_bounds__(64)
void phasor_kernel(const float* __restrict__ Q, const float* __restrict__ R,
                   const float* __restrict__ phi_w, const float* __restrict__ phi_b,
                   bf16_t* __restrict__ Qri, bf16_t* __restrict__ Rri,
                   int S, int H, int HD) {
    const int idx = blockIdx.x;
    const int h = idx % H;
    const int rem = idx / H;
    const int s = rem % S;
    const int b = rem / S;
    const int e = threadIdx.x;

    __shared__ float qrow[64];
    __shared__ float rrow[64];
    const size_t rowoff = ((size_t)b * S + s) * (size_t)(H * HD) + (size_t)h * HD;
    qrow[e] = Q[rowoff + e];
    rrow[e] = R[rowoff + e];
    __syncthreads();

    const float* wr = phi_w + (size_t)e * HD;
    float pq = phi_b[e], pr = phi_b[e];
#pragma unroll 8
    for (int d = 0; d < 64; ++d) {
        const float w = wr[d];
        pq += w * qrow[d];
        pr += w * rrow[d];
    }
    float sq, cq, sr, cr;
    __sincosf(pq, &sq, &cq);
    __sincosf(pr, &sr, &cr);
    const float qv = qrow[e], rv = rrow[e];
    const size_t obase = (((size_t)b * H + h) * S + s) * (size_t)(2 * HD);
    Qri[obase + e]      = (bf16_t)(qv * cq);
    Qri[obase + HD + e] = (bf16_t)(qv * sq);
    Rri[obase + e]      = (bf16_t)(rv * cr);
    Rri[obase + HD + e] = (bf16_t)(rv * sr);
}

// ---------------------------------------------------------------------------
// Fused attention: per (b,h), per 64-q tile: stream k in steps of 32,
//   S_tile = Qri @ Rri^T (K=128, bf16 WMMA), sumsq += S^2,
//   O += S_tile @ Hm_tile (score tile bounced through per-wave LDS),
//   final: O *= 1/max(sqrt(sumsq),1e-12), write bf16 into [B*S, D].
// Hm arrives pre-transposed ([B,H,HD,S]) so both stages are pure B128 async
// copies; tiles are double-buffered. SxS scores never touch HBM (~268 MB saved).
// ---------------------------------------------------------------------------
__global__ __launch_bounds__(128)
void attn_fused_kernel(const bf16_t* __restrict__ Qri, const bf16_t* __restrict__ Rri,
                       const bf16_t* __restrict__ HmT_g, bf16_t* __restrict__ O,
                       int S, int H, int HD) {
    const int bh = blockIdx.y;      // b*H + h
    const int q0 = blockIdx.x * 64;
    const int tid = threadIdx.x, wave = tid >> 5, lane = tid & 31;
    const int half = lane >> 4, ln = lane & 15;

    __shared__ __align__(16) bf16_t Rs[2][32][128];   // Rri k-tiles  (16 KB)
    __shared__ __align__(16) bf16_t HmT[2][64][32];   // HmT k-tiles  (8 KB)
    __shared__ __align__(16) bf16_t Ss[4][16][32];    // score bounce (4 KB)

    const bf16_t* qbase  = Qri   + (size_t)bh * S * 128;
    const bf16_t* rbase  = Rri   + (size_t)bh * S * 128;
    const bf16_t* htbase = HmT_g + (size_t)bh * HD * S;   // [d][s] rows

    auto stageKT = [&](int bb, int k0) {
#pragma unroll
        for (int i = 0; i < 4; ++i) {   // Rri: 32x128 bf16 = 512 x 16B
            int c = tid + i * 128;
            cp_b128(&((uint4*)&Rs[bb][0][0])[c],
                    (const uint4*)(rbase + (size_t)k0 * 128) + c);
        }
#pragma unroll
        for (int i = 0; i < 2; ++i) {   // HmT: 64 d-rows x 32 k = 256 x 16B
            int c = tid + i * 128;
            int d = c >> 2, cc = c & 3;
            cp_b128(&HmT[bb][d][cc * 8], htbase + (size_t)d * S + k0 + cc * 8);
        }
    };

    // Q fragments for this wave's 16 rows over the whole K=128 contraction.
    v16bf aq[4];
    {
        const int m = q0 + wave * 16 + ln;
        const bf16_t* qr = qbase + (size_t)m * 128;
        const int b0 = half ? 8 : 0;
#pragma unroll
        for (int kc = 0; kc < 4; ++kc) {
            v8bf lo = *(const v8bf*)(qr + kc * 32 + b0);
            v8bf hi = *(const v8bf*)(qr + kc * 32 + b0 + 16);
            aq[kc] = make_frag(lo, hi);
        }
    }

    v8f acc[4] = {};
    float ssq[8] = {};

    stageKT(0, 0);
    wait_async();
    __syncthreads();

    int cur = 0;
    for (int k0 = 0; k0 < S; k0 += 32) {
        if (k0 + 32 < S) stageKT(cur ^ 1, k0 + 32);   // overlap next k-slab

        // --- scores: two 16-wide k column tiles, K=128 contraction ---
        v8f sacc[2] = {};
#pragma unroll
        for (int t = 0; t < 2; ++t) {
            const int nl = t * 16 + ln;
#pragma unroll
            for (int kc = 0; kc < 4; ++kc) {
                const int kk0 = kc * 32 + (half ? 8 : 0);
                v8bf blo = *(const v8bf*)&Rs[cur][nl][kk0];
                v8bf bhi = *(const v8bf*)&Rs[cur][nl][kk0 + 16];
                sacc[t] = wmma_bf16(aq[kc], make_frag(blo, bhi), sacc[t]);
            }
        }

        // sumsq + bounce scores through per-wave LDS (D-layout -> A-layout).
#pragma unroll
        for (int t = 0; t < 2; ++t) {
#pragma unroll
            for (int r = 0; r < 8; ++r) {
                const float v = sacc[t][r];
                ssq[r] += v * v;
                Ss[wave][half ? 8 + r : r][t * 16 + ln] = (bf16_t)v;
            }
        }
        v16bf sa;
        {
            const int b0 = half ? 8 : 0;
            v8bf lo = *(const v8bf*)&Ss[wave][ln][b0];
            v8bf hi = *(const v8bf*)&Ss[wave][ln][b0 + 16];
            sa = make_frag(lo, hi);
        }

        // --- apply: O(16x64) += S(16x32) @ Hm(32x64) ---
#pragma unroll
        for (int t = 0; t < 4; ++t) {
            const int d = t * 16 + ln;
            const int kk0 = half ? 8 : 0;
            v8bf blo = *(const v8bf*)&HmT[cur][d][kk0];
            v8bf bhi = *(const v8bf*)&HmT[cur][d][kk0 + 16];
            acc[t] = wmma_bf16(sa, make_frag(blo, bhi), acc[t]);
        }

        wait_async();
        __syncthreads();
        cur ^= 1;
    }

    // Row sum-of-squares: reduce across the 16-lane column group.
#pragma unroll
    for (int r = 0; r < 8; ++r) {
        float v = ssq[r];
        v += __shfl_xor(v, 1, 32);
        v += __shfl_xor(v, 2, 32);
        v += __shfl_xor(v, 4, 32);
        v += __shfl_xor(v, 8, 32);
        ssq[r] = v;
    }

    const int b = bh / H, hh = bh % H;
    const int HD_all = H * HD;
#pragma unroll
    for (int r = 0; r < 8; ++r) {
        const float scale = 1.0f / fmaxf(sqrtf(ssq[r]), 1e-12f);
        const int m = q0 + wave * 16 + (half ? 8 + r : r);
        const size_t rowo = ((size_t)b * S + m) * HD_all + (size_t)hh * HD;
#pragma unroll
        for (int t = 0; t < 4; ++t)
            O[rowo + t * 16 + ln] = (bf16_t)(acc[t][r] * scale);
    }
}

// ---------------------------------------------------------------------------
// Host-side launch sequence.
// ---------------------------------------------------------------------------
extern "C" void kernel_launch(void* const* d_in, const int* in_sizes, int n_in,
                              void* d_out, int out_size, void* d_ws, size_t ws_size,
                              hipStream_t stream) {
    (void)in_sizes; (void)n_in; (void)out_size; (void)ws_size;

    const float* x     = (const float*)d_in[0];
    const float* ln_g  = (const float*)d_in[1];
    const float* ln_b  = (const float*)d_in[2];
    const float* z_w   = (const float*)d_in[3];
    const float* z_b   = (const float*)d_in[4];
    const float* zn_g  = (const float*)d_in[5];
    const float* zn_b  = (const float*)d_in[6];
    const float* q_w   = (const float*)d_in[7];
    const float* q_b   = (const float*)d_in[8];
    const float* r_w   = (const float*)d_in[9];
    const float* r_b   = (const float*)d_in[10];
    const float* h_w   = (const float*)d_in[11];
    const float* h_b   = (const float*)d_in[12];
    const float* phi_w = (const float*)d_in[13];
    const float* phi_b = (const float*)d_in[14];
    const float* o_w   = (const float*)d_in[15];
    const float* o_b   = (const float*)d_in[16];
    const float* ls    = (const float*)d_in[17];

    const int B = 4, S = 1024, D = 1024, H = 16, HD = 64;
    const int M = B * S, N = D, K = D;

    char* ws = (char*)d_ws;
    const size_t MB = 1ull << 20;
    bf16_t* zw_bf  = (bf16_t*)(ws + 0 * MB);
    bf16_t* qw_bf  = (bf16_t*)(ws + 2 * MB);
    bf16_t* rw_bf  = (bf16_t*)(ws + 4 * MB);
    bf16_t* hw_bf  = (bf16_t*)(ws + 6 * MB);
    bf16_t* ow_bf  = (bf16_t*)(ws + 8 * MB);
    bf16_t* xn_bf  = (bf16_t*)(ws + 10 * MB);  // 8 MB
    bf16_t* Z_bf   = (bf16_t*)(ws + 18 * MB);  // 8 MB
    bf16_t* Hm_bf  = (bf16_t*)(ws + 26 * MB);  // 8 MB [B,H,HD,S] (d-major)
    bf16_t* Qri    = (bf16_t*)(ws + 34 * MB);  // 16 MB [B,H,S,128]
    bf16_t* Rri    = (bf16_t*)(ws + 50 * MB);  // 16 MB
    bf16_t* Ob_bf  = (bf16_t*)(ws + 66 * MB);  // 8 MB [B*S, D]
    float*  F1     = (float*)(ws + 74 * MB);   // 16 MB (Zpre, then Q)
    float*  F2     = (float*)(ws + 90 * MB);   // 16 MB (R)

    const int nW = D * D;
    dim3 cgrd((nW + 255) / 256), cblk(256);
    hipLaunchKernelGGL(cvt_f32_bf16, cgrd, cblk, 0, stream, z_w, zw_bf, nW);
    hipLaunchKernelGGL(cvt_f32_bf16, cgrd, cblk, 0, stream, q_w, qw_bf, nW);
    hipLaunchKernelGGL(cvt_f32_bf16, cgrd, cblk, 0, stream, r_w, rw_bf, nW);
    hipLaunchKernelGGL(cvt_f32_bf16, cgrd, cblk, 0, stream, h_w, hw_bf, nW);
    hipLaunchKernelGGL(cvt_f32_bf16, cgrd, cblk, 0, stream, o_w, ow_bf, nW);

    // 1. xn = LN(x)
    hipLaunchKernelGGL(layernorm_f32_bf16, dim3(M), dim3(128), 0, stream,
                       x, ln_g, ln_b, xn_bf, D);

    dim3 ggrd(N / 64, M / 64), gblk(128);
    // 2. Zpre = xn @ z_w^T + z_b   (fp32)
    hipLaunchKernelGGL((gemm_wmma_bf16<0>), ggrd, gblk, 0, stream,
                       xn_bf, zw_bf, z_b, (void*)F1, nullptr, nullptr,
                       M, N, K, S, H, HD);
    // 3. Z = LN(Zpre)
    hipLaunchKernelGGL(layernorm_f32_bf16, dim3(M), dim3(128), 0, stream,
                       F1, zn_g, zn_b, Z_bf, D);
    // 4. Q = Z@q_w^T+q_b (fp32, reuse F1), R = Z@r_w^T+r_b (fp32)
    hipLaunchKernelGGL((gemm_wmma_bf16<0>), ggrd, gblk, 0, stream,
                       Z_bf, qw_bf, q_b, (void*)F1, nullptr, nullptr,
                       M, N, K, S, H, HD);
    hipLaunchKernelGGL((gemm_wmma_bf16<0>), ggrd, gblk, 0, stream,
                       Z_bf, rw_bf, r_b, (void*)F2, nullptr, nullptr,
                       M, N, K, S, H, HD);
    // 5. Hm = Z@h_w^T+h_b, straight to bf16 d-major [B,H,HD,S]
    hipLaunchKernelGGL((gemm_wmma_bf16<1>), ggrd, gblk, 0, stream,
                       Z_bf, hw_bf, h_b, (void*)Hm_bf, nullptr, nullptr,
                       M, N, K, S, H, HD);
    // 6. phasor: Qri/Rri bf16 [B,H,S,2*HD]
    hipLaunchKernelGGL(phasor_kernel, dim3(B * S * H), dim3(64), 0, stream,
                       F1, F2, phi_w, phi_b, Qri, Rri, S, H, HD);
    // 7. fused scores + L2-norm + apply -> Ob_bf [B*S, D]
    hipLaunchKernelGGL(attn_fused_kernel, dim3(S / 64, B * H), dim3(128), 0, stream,
                       Qri, Rri, Hm_bf, Ob_bf, S, H, HD);
    // 8. out = x + ls * (Ob @ o_w^T + o_b)
    hipLaunchKernelGGL((gemm_wmma_bf16<2>), ggrd, gblk, 0, stream,
                       Ob_bf, ow_bf, o_b, d_out, x, ls,
                       M, N, K, S, H, HD);
}
